// QVMetaLA_self_aug_39986145525841
// MI455X (gfx1250) — matrix operand order repaired and checked by reading, not compiled, not run on hardware
//
#include <hip/hip_runtime.h>
#include <cstdint>
#include <cstddef>

// ---------------------------------------------------------------------------
// QVMetaLA (gated linear attention block) for MI455X / gfx1250, wave32 + WMMA.
//   1) pack weights f32 -> bf16 transposed (N x K)
//   2) causal depthwise conv(W=4) + SiLU -> xc (bf16)
//   3) WMMA GEMM: xc[8192x1024] @ W[1024x3072] -> q|kg|v|g (f32)
//   4) gk = logsigmoid(kg)/16, k = 1-exp(gk); g += bg
//   5) chunk-parallel GLA (C=64), 3 passes:
//        p1 (1024 blks): A=q~@k^T masked, o_intra=A@vT, S_loc=k-T@v, e^Bc, q~
//        p2 (32 blks)  : serial prefix S0[ch] = run; run = run*e^Bc + S_loc
//        p3 (1024 blks): o += q~ @ S0^T  (WMMA)
//   6) aug-sigmoid + LayerNorm(head) + SiLU(g) gate -> pre (bf16)
//   7) WMMA GEMM: pre[8192x1024] @ wout[1024x1024] -> out (f32)
// LDS staging uses GLOBAL_LOAD_ASYNC_TO_LDS_B128 (ASYNCcnt) when the builtin
// is available; falls back to VGPR-bounced copies otherwise.
// ---------------------------------------------------------------------------

typedef __attribute__((ext_vector_type(16))) __bf16 bf16x16;
typedef __attribute__((ext_vector_type(8)))  float  f32x8;
typedef int v4i_ __attribute__((vector_size(4 * sizeof(int))));

#define B_    4
#define L_    2048
#define E_    1024
#define DK_   512
#define H_    8
#define DKH_  64
#define DV_   128
#define M_    (B_ * L_)     // 8192 rows
#define NALL_ 3072          // q(512) | kg(512) | v(1024) | g(1024)
#define SCALE_ 0.125f       // dkh^-0.5 = 1/8
#define LN_EPS_ 1e-5f
#define CHUNK_ 64
#define NCHUNK_ (L_ / CHUNK_)          // 32
#define NBH_ (B_ * H_)                 // 32
#define NBHC_ (NBH_ * NCHUNK_)         // 1024
#define PAD_  72            // LDS row stride (ushorts): 144B, 16B-aligned, debanked

struct __align__(16) U128 { unsigned int x, y, z, w; };
union AFragU { bf16x16 v; U128 q[2]; };

#if __has_builtin(__builtin_amdgcn_global_load_async_to_lds_b128) && \
    __has_builtin(__builtin_amdgcn_s_wait_asynccnt)
#define USE_ASYNC_LDS 1
#else
#define USE_ASYNC_LDS 0
#endif

#if USE_ASYNC_LDS
// 16B global->LDS DMA, no VGPR bounce; tracked by ASYNCcnt.
// Prototype (from clang diagnostic): (v4i32 AS1*, v4i32 AS3*, imm off, imm cpol)
__device__ __forceinline__ void cp16_async(void* lds, const void* g) {
  __builtin_amdgcn_global_load_async_to_lds_b128(
      (__attribute__((address_space(1))) v4i_*)g,
      (__attribute__((address_space(3))) v4i_*)lds, 0, 0);
}
__device__ __forceinline__ void async_wait0() {
  __builtin_amdgcn_s_wait_asynccnt(0);
}
#else
__device__ __forceinline__ void cp16_async(void* lds, const void* g) {
  *(U128*)lds = *(const U128*)g;
}
__device__ __forceinline__ void async_wait0() {}
#endif

__device__ __forceinline__ unsigned short f32_to_bf16(float f) {
  union { float f; unsigned int u; } v; v.f = f;
  unsigned int r = v.u + 0x7FFFu + ((v.u >> 16) & 1u);  // round-nearest-even
  return (unsigned short)(r >> 16);
}
__device__ __forceinline__ float sigmoidf_(float x) { return 1.f / (1.f + __expf(-x)); }
__device__ __forceinline__ float siluf_(float x)    { return x * sigmoidf_(x); }
__device__ __forceinline__ float log_sigmoidf_(float x) {
  float l = __logf(1.f + __expf(-fabsf(x)));
  return (x >= 0.f) ? -l : (x - l);
}

// WMMA fragment loaders (CDNA5 ISA 7.12.2 lane layouts).
// A 16x32: lane m=l%16, K chunks [kb,kb+8) & [kb+16,kb+24), kb=(l>=16)?8:0
__device__ __forceinline__ bf16x16 loadA_frag(const unsigned short* base, int row0,
                                              int stride, int k0, int lane) {
  AFragU f;
  const unsigned short* p = base + (row0 + (lane & 15)) * stride + k0 + ((lane & 16) >> 1);
  f.q[0] = *(const U128*)p;
  f.q[1] = *(const U128*)(p + 16);
  return f.v;
}
// B 32x16 (from Bt[n][k]): lane n=l%16, K chunks [kb,kb+8) & [kb+16,kb+24), kb=(l>=16)?8:0
__device__ __forceinline__ bf16x16 loadB_frag(const unsigned short* base, int col0,
                                              int stride, int k0, int lane) {
  AFragU f;
  const unsigned short* p = base + (col0 + (lane & 15)) * stride + k0 + ((lane & 16) >> 1);
  f.q[0] = *(const U128*)p;
  f.q[1] = *(const U128*)(p + 16);
  return f.v;
}

// ---------------------------------------------------------------------------
// Weight packing: f32 [K][Ncols] -> bf16 transposed [N][K]
// ---------------------------------------------------------------------------
__global__ __launch_bounds__(256) void pack_wall_kernel(
    const float* __restrict__ wq, const float* __restrict__ wkg,
    const float* __restrict__ wv, const float* __restrict__ wg,
    unsigned short* __restrict__ wt) {
  int idx = blockIdx.x * 256 + threadIdx.x;          // over NALL_*E_
  int n  = idx >> 10;
  int kk = idx & 1023;
  const float* src; int nl, cols;
  if (n < 512)       { src = wq;  nl = n;        cols = 512;  }
  else if (n < 1024) { src = wkg; nl = n - 512;  cols = 512;  }
  else if (n < 2048) { src = wv;  nl = n - 1024; cols = 1024; }
  else               { src = wg;  nl = n - 2048; cols = 1024; }
  wt[idx] = f32_to_bf16(src[(size_t)kk * cols + nl]);
}

__global__ __launch_bounds__(256) void pack_wout_kernel(
    const float* __restrict__ wout, unsigned short* __restrict__ wt) {
  int idx = blockIdx.x * 256 + threadIdx.x;          // over E_*E_
  int n = idx >> 10, kk = idx & 1023;
  wt[idx] = f32_to_bf16(wout[(size_t)kk * E_ + n]);
}

// ---------------------------------------------------------------------------
// Causal depthwise conv (W=4) + SiLU, output bf16
// ---------------------------------------------------------------------------
__global__ __launch_bounds__(256) void conv_silu_kernel(
    const float* __restrict__ x, const float* __restrict__ cw,
    unsigned short* __restrict__ xcb) {
  int idx = blockIdx.x * 256 + threadIdx.x;          // (b*L + l)*E + e
  int e = idx & (E_ - 1);
  int bl = idx >> 10;
  int l = bl & (L_ - 1);
  float acc = 0.f;
#pragma unroll
  for (int w = 0; w < 4; ++w) {
    int ls = l + w - 3;
    if (ls >= 0) acc += x[(size_t)(bl + w - 3) * E_ + e] * cw[e * 4 + w];
  }
  xcb[idx] = f32_to_bf16(siluf_(acc));
}

// ---------------------------------------------------------------------------
// WMMA bf16 GEMM: C[M,N] = A[M,K] @ B[K,N]; B supplied transposed Bt[N][K].
// Block tile 128(M) x 64(N), K-step 32. 8 waves in 4x2, each wave: 2x2 16x16.
// ---------------------------------------------------------------------------
__global__ __launch_bounds__(256) void gemm_bf16_wmma(
    const unsigned short* __restrict__ A, const unsigned short* __restrict__ Bt,
    float* __restrict__ C, int M, int N, int K) {
  __shared__ __align__(16) unsigned short sA[128][32];
  __shared__ __align__(16) unsigned short sB[64][32];
  const int tid  = threadIdx.x;
  const int lane = tid & 31;
  const int wave = tid >> 5;
  const int wm = wave & 3;
  const int wn = wave >> 2;
  const int m0 = blockIdx.y * 128;
  const int n0 = blockIdx.x * 64;

  f32x8 acc[2][2];
#pragma unroll
  for (int i = 0; i < 2; ++i)
#pragma unroll
    for (int j = 0; j < 2; ++j)
      acc[i][j] = (f32x8){0.f, 0.f, 0.f, 0.f, 0.f, 0.f, 0.f, 0.f};

  const int aChunk = tid * 2;
  const int aRow   = aChunk >> 2;
  const int aCol   = (aChunk & 3) * 8;
  const int bRow   = tid >> 2;
  const int bCol   = (tid & 3) * 8;

  for (int kt = 0; kt < K; kt += 32) {
    const unsigned short* gA = A + (size_t)(m0 + aRow) * K + kt + aCol;
    cp16_async(&sA[aRow][aCol],     gA);
    cp16_async(&sA[aRow][aCol + 8], gA + 8);
    cp16_async(&sB[bRow][bCol], Bt + (size_t)(n0 + bRow) * K + kt + bCol);
    if (kt + 32 < K) {
      __builtin_prefetch(A + (size_t)(m0 + aRow) * K + kt + 32 + aCol, 0, 1);
      __builtin_prefetch(Bt + (size_t)(n0 + bRow) * K + kt + 32 + bCol, 0, 1);
    }
    async_wait0();
    __syncthreads();

    AFragU a[2], b[2];
#pragma unroll
    for (int i = 0; i < 2; ++i) {
      int row = wm * 32 + i * 16 + (lane & 15);
      a[i].q[0] = *(const U128*)&sA[row][(lane & 16) >> 1];
      a[i].q[1] = *(const U128*)&sA[row][((lane & 16) >> 1) + 16];
    }
#pragma unroll
    for (int j = 0; j < 2; ++j) {
      int col = wn * 32 + j * 16 + (lane & 15);
      b[j].q[0] = *(const U128*)&sB[col][(lane & 16) >> 1];
      b[j].q[1] = *(const U128*)&sB[col][((lane & 16) >> 1) + 16];
    }
#pragma unroll
    for (int i = 0; i < 2; ++i)
#pragma unroll
      for (int j = 0; j < 2; ++j)
        acc[i][j] = __builtin_amdgcn_wmma_f32_16x16x32_bf16(
            false, a[i].v, false, b[j].v, (short)0, acc[i][j], false, false);
    __syncthreads();
  }

  const int cm = (lane >> 4) * 8;
  const int cn = lane & 15;
#pragma unroll
  for (int i = 0; i < 2; ++i)
#pragma unroll
    for (int j = 0; j < 2; ++j) {
      int rbase = m0 + wm * 32 + i * 16 + cm;
      int cbase = n0 + wn * 32 + j * 16 + cn;
#pragma unroll
      for (int r = 0; r < 8; ++r)
        C[(size_t)(rbase + r) * N + cbase] = acc[i][j][r];
    }
}

// ---------------------------------------------------------------------------
// gk = logsigmoid(kg)/16 ; k = 1 - exp(gk) ; g += bg
// ---------------------------------------------------------------------------
__global__ __launch_bounds__(256) void gk_kernel(
    const float* __restrict__ qkvg, float* __restrict__ gkb,
    float* __restrict__ kbuf) {
  int idx = blockIdx.x * 256 + threadIdx.x;          // over M_*DK_
  int row = idx >> 9, cc = idx & 511;
  float kg = qkvg[(size_t)row * NALL_ + 512 + cc];
  float gk = log_sigmoidf_(kg) * (1.f / 16.f);
  gkb[idx]  = gk;
  kbuf[idx] = 1.f - __expf(gk);
}

__global__ __launch_bounds__(256) void gbias_kernel(
    float* __restrict__ qkvg, const float* __restrict__ bg) {
  int idx = blockIdx.x * 256 + threadIdx.x;          // over M_*E_
  int row = idx >> 10, cc = idx & 1023;
  qkvg[(size_t)row * NALL_ + 2048 + cc] += bg[cc];
}

// ---------------------------------------------------------------------------
// GLA pass 1: fully parallel over (b,h,chunk) = 1024 blocks, 8 waves each.
// ---------------------------------------------------------------------------
__global__ __launch_bounds__(256) void gla_pass1_kernel(
    const float* __restrict__ qkvg, const float* __restrict__ gkb,
    const float* __restrict__ kbuf, float* __restrict__ obuf,
    unsigned short* __restrict__ qt_g, float* __restrict__ slocal,
    float* __restrict__ expbc) {
  __shared__ float gkc[64][65];
  __shared__ __align__(16) unsigned short qt_bf[64][PAD_];   // q~  [t][d]
  __shared__ __align__(16) unsigned short kh_bf[64][PAD_];   // k^  [s][d]
  __shared__ __align__(16) unsigned short kbT_bf[64][PAD_];  // k-T [d][s]
  __shared__ __align__(16) unsigned short vT_bf[128][PAD_];  // v^T [j][s]
  __shared__ __align__(16) unsigned short A_bf[64][PAD_];    // A   [t][s]

  const int bhc = blockIdx.x;              // bh*NCHUNK_ + ch
  const int ch = bhc & (NCHUNK_ - 1);
  const int bh = bhc >> 5;
  const int b = bh >> 3, h = bh & 7;
  const int tid = threadIdx.x;
  const int lane = tid & 31;
  const int wave = tid >> 5;
  const size_t row0 = (size_t)b * L_ + ch * CHUNK_;
  const f32x8 zero8 = (f32x8){0.f, 0.f, 0.f, 0.f, 0.f, 0.f, 0.f, 0.f};

  for (int i = tid; i < 64 * 64; i += 256) {
    int t = i >> 6, d = i & 63;
    gkc[t][d] = gkb[(row0 + t) * DK_ + h * 64 + d];
  }
  __syncthreads();
  if (tid < 64) {
    float run = 0.f;
    for (int t = 0; t < 64; ++t) { run += gkc[t][tid]; gkc[t][tid] = run; }
    expbc[(size_t)bhc * 64 + tid] = __expf(run);
  }
  __syncthreads();
  for (int i = tid; i < 64 * 64; i += 256) {
    int t = i >> 6, d = i & 63;
    float Bt = gkc[t][d];
    float q = qkvg[(row0 + t) * NALL_ + h * 64 + d];
    unsigned short qb = f32_to_bf16(q * __expf(Bt) * SCALE_);
    qt_bf[t][d] = qb;
    qt_g[(size_t)bhc * 4096 + t * 64 + d] = qb;
    float k = kbuf[(row0 + t) * DK_ + h * 64 + d];
    kh_bf[t][d]  = f32_to_bf16(k * __expf(-Bt));
    kbT_bf[d][t] = f32_to_bf16(k * __expf(gkc[63][d] - Bt));
  }
  for (int i = tid; i < 64 * 128; i += 256) {
    int s = i >> 7, j = i & 127;
    vT_bf[j][s] = f32_to_bf16(qkvg[(row0 + s) * NALL_ + 1024 + h * 128 + j]);
  }
  __syncthreads();

  // A = q~ @ k^T : 16 tiles, 2 per wave, causal mask, requantize
#pragma unroll
  for (int tt = 0; tt < 2; ++tt) {
    int tile = wave * 2 + tt;
    int ti = tile >> 2, si = tile & 3;
    f32x8 acc = zero8;
#pragma unroll
    for (int ks = 0; ks < 64; ks += 32)
      acc = __builtin_amdgcn_wmma_f32_16x16x32_bf16(
          false, loadA_frag(&qt_bf[0][0], ti * 16, PAD_, ks, lane),
          false, loadB_frag(&kh_bf[0][0], si * 16, PAD_, ks, lane),
          (short)0, acc, false, false);
    int m0 = (lane >> 4) << 3, n = lane & 15;
#pragma unroll
    for (int r = 0; r < 8; ++r) {
      int t = ti * 16 + m0 + r, s = si * 16 + n;
      A_bf[t][s] = (s <= t) ? f32_to_bf16(acc[r]) : (unsigned short)0;
    }
  }
  __syncthreads();

  // o_intra = A @ vT : 32 tiles, 4 per wave -> obuf
  {
    int ti = wave & 3;
#pragma unroll
    for (int jj = 0; jj < 4; ++jj) {
      int ji = (wave >> 2) * 4 + jj;
      f32x8 acc = zero8;
#pragma unroll
      for (int ks = 0; ks < 64; ks += 32)
        acc = __builtin_amdgcn_wmma_f32_16x16x32_bf16(
            false, loadA_frag(&A_bf[0][0], ti * 16, PAD_, ks, lane),
            false, loadB_frag(&vT_bf[0][0], ji * 16, PAD_, ks, lane),
            (short)0, acc, false, false);
      int m0 = (lane >> 4) << 3, n = lane & 15;
#pragma unroll
      for (int r = 0; r < 8; ++r) {
        int t = ti * 16 + m0 + r, j = ji * 16 + n;
        obuf[(row0 + t) * E_ + h * DV_ + j] = acc[r];
      }
    }
  }

  // S_loc = k-T @ vT : 32 tiles, 4 per wave -> slocal [d][j]
  {
    int di = wave & 3;
#pragma unroll
    for (int jj = 0; jj < 4; ++jj) {
      int ji = (wave >> 2) * 4 + jj;
      f32x8 acc = zero8;
#pragma unroll
      for (int ks = 0; ks < 64; ks += 32)
        acc = __builtin_amdgcn_wmma_f32_16x16x32_bf16(
            false, loadA_frag(&kbT_bf[0][0], di * 16, PAD_, ks, lane),
            false, loadB_frag(&vT_bf[0][0], ji * 16, PAD_, ks, lane),
            (short)0, acc, false, false);
      int m0 = (lane >> 4) << 3, n = lane & 15;
#pragma unroll
      for (int r = 0; r < 8; ++r) {
        int d = di * 16 + m0 + r, j = ji * 16 + n;
        slocal[(size_t)bhc * 8192 + d * 128 + j] = acc[r];
      }
    }
  }
}

// ---------------------------------------------------------------------------
// GLA pass 2: serial prefix over chunks (elementwise). 32 blocks, 256 threads.
// ---------------------------------------------------------------------------
__global__ __launch_bounds__(256) void gla_pass2_kernel(
    const float* __restrict__ slocal, const float* __restrict__ expbc,
    unsigned short* __restrict__ s0t_g) {
  const int bh = blockIdx.x;
  const int tid = threadIdx.x;
  const int d = tid & 63;
  const int j0 = (tid >> 6) * 32;
  float run[32];
#pragma unroll
  for (int jj = 0; jj < 32; ++jj) run[jj] = 0.f;
  for (int ch = 0; ch < NCHUNK_; ++ch) {
    const size_t base = (size_t)(bh * NCHUNK_ + ch);
    float e = expbc[base * 64 + d];
#pragma unroll
    for (int jj = 0; jj < 32; ++jj) {
      int j = j0 + jj;
      s0t_g[base * 8192 + j * 64 + d] = f32_to_bf16(run[jj]);
      run[jj] = fmaf(run[jj], e, slocal[base * 8192 + (size_t)d * 128 + j]);
    }
  }
}

// ---------------------------------------------------------------------------
// GLA pass 3: o += q~ @ S0^T  (1024 blocks, 32 tiles, 4 per wave)
// Tile staging through the async-LDS path (16B granules into padded rows).
// ---------------------------------------------------------------------------
__global__ __launch_bounds__(256) void gla_pass3_kernel(
    const unsigned short* __restrict__ qt_g,
    const unsigned short* __restrict__ s0t_g, float* __restrict__ obuf) {
  __shared__ __align__(16) unsigned short qt_bf[64][PAD_];
  __shared__ __align__(16) unsigned short s0_bf[128][PAD_];
  const int bhc = blockIdx.x;
  const int ch = bhc & (NCHUNK_ - 1);
  const int bh = bhc >> 5;
  const int b = bh >> 3, h = bh & 7;
  const int tid = threadIdx.x;
  const int lane = tid & 31;
  const int wave = tid >> 5;
  const size_t row0 = (size_t)b * L_ + ch * CHUNK_;
  const f32x8 zero8 = (f32x8){0.f, 0.f, 0.f, 0.f, 0.f, 0.f, 0.f, 0.f};

  for (int i = tid; i < 512; i += 256) {    // 64x64 ushorts = 512 x 16B
    int r = i >> 3, c = (i & 7) * 8;
    cp16_async(&qt_bf[r][c], qt_g + (size_t)bhc * 4096 + i * 8);
  }
  for (int i = tid; i < 1024; i += 256) {   // 128x64 ushorts = 1024 x 16B
    int r = i >> 3, c = (i & 7) * 8;
    cp16_async(&s0_bf[r][c], s0t_g + (size_t)bhc * 8192 + i * 8);
  }
  async_wait0();
  __syncthreads();

  int ti = wave & 3;
#pragma unroll
  for (int jj = 0; jj < 4; ++jj) {
    int ji = (wave >> 2) * 4 + jj;
    f32x8 acc = zero8;
#pragma unroll
    for (int ks = 0; ks < 64; ks += 32)
      acc = __builtin_amdgcn_wmma_f32_16x16x32_bf16(
          false, loadA_frag(&qt_bf[0][0], ti * 16, PAD_, ks, lane),
          false, loadB_frag(&s0_bf[0][0], ji * 16, PAD_, ks, lane),
          (short)0, acc, false, false);
    int m0 = (lane >> 4) << 3, n = lane & 15;
#pragma unroll
    for (int r = 0; r < 8; ++r) {
      int t = ti * 16 + m0 + r, j = ji * 16 + n;
      size_t oi = (row0 + t) * E_ + h * DV_ + j;
      obuf[oi] = obuf[oi] + acc[r];
    }
  }
}

// ---------------------------------------------------------------------------
// Epilogue: o += sigmoid(aug_w * v); LayerNorm over head_dim; pre = silu(g)*o
// ---------------------------------------------------------------------------
__global__ __launch_bounds__(256) void post_kernel(
    const float* __restrict__ qkvg, const float* __restrict__ kbuf,
    const float* __restrict__ obuf, const float* __restrict__ aug,
    unsigned short* __restrict__ preb) {
  __shared__ float r1[256], r2[256];
  const size_t row = blockIdx.x;
  const int tid = threadIdx.x;
  const int h   = tid >> 5;
  const int g32 = tid & 31;
  const size_t b3 = row * NALL_;

  int d = g32 * 2;
  float pa =
      qkvg[b3 + h * 64 + d]     * aug[h * 64 + d]     * kbuf[row * DK_ + h * 64 + d] +
      qkvg[b3 + h * 64 + d + 1] * aug[h * 64 + d + 1] * kbuf[row * DK_ + h * 64 + d + 1];
  r1[tid] = pa; __syncthreads();
  for (int s = 16; s >= 1; s >>= 1) {
    if (g32 < s) r1[tid] += r1[tid + s];
    __syncthreads();
  }
  float augw = r1[h * 32];
  __syncthreads();

  float ov[4], s1 = 0.f, s2 = 0.f;
#pragma unroll
  for (int j = 0; j < 4; ++j) {
    int gc = tid * 4 + j;
    float v = qkvg[b3 + 1024 + gc];
    float o = obuf[row * E_ + gc] + sigmoidf_(augw * v);
    ov[j] = o; s1 += o; s2 += o * o;
  }
  r1[tid] = s1; r2[tid] = s2; __syncthreads();
  for (int s = 16; s >= 1; s >>= 1) {
    if (g32 < s) { r1[tid] += r1[tid + s]; r2[tid] += r2[tid + s]; }
    __syncthreads();
  }
  float mu   = r1[h * 32] * (1.f / 128.f);
  float var  = r2[h * 32] * (1.f / 128.f) - mu * mu;
  float rstd = rsqrtf(var + LN_EPS_);
#pragma unroll
  for (int j = 0; j < 4; ++j) {
    int gc = tid * 4 + j;
    float g = qkvg[b3 + 2048 + gc];
    preb[row * E_ + gc] = f32_to_bf16((ov[j] - mu) * rstd * siluf_(g));
  }
}

// ---------------------------------------------------------------------------
extern "C" void kernel_launch(void* const* d_in, const int* in_sizes, int n_in,
                              void* d_out, int out_size, void* d_ws, size_t ws_size,
                              hipStream_t stream) {
  (void)in_sizes; (void)n_in; (void)out_size; (void)ws_size;
  const float* x    = (const float*)d_in[0];
  const float* cw   = (const float*)d_in[1];
  const float* wq   = (const float*)d_in[2];
  const float* wkg  = (const float*)d_in[3];
  const float* wv   = (const float*)d_in[4];
  const float* wg   = (const float*)d_in[5];
  const float* bg   = (const float*)d_in[6];
  const float* wout = (const float*)d_in[7];
  const float* aug  = (const float*)d_in[8];
  float* out = (float*)d_out;

  char* ws = (char*)d_ws;
  size_t off = 0;
  auto alloc = [&](size_t bytes) -> void* {
    void* p = ws + off;
    off += (bytes + 255) & ~(size_t)255;
    return p;
  };
  unsigned short* xcb   = (unsigned short*)alloc((size_t)M_ * E_ * 2);
  unsigned short* wallt = (unsigned short*)alloc((size_t)NALL_ * E_ * 2);
  unsigned short* woutt = (unsigned short*)alloc((size_t)E_ * E_ * 2);
  unsigned short* preb  = (unsigned short*)alloc((size_t)M_ * E_ * 2);
  float* qkvg = (float*)alloc((size_t)M_ * NALL_ * 4);
  float* gkb  = (float*)alloc((size_t)M_ * DK_ * 4);
  float* kbuf = (float*)alloc((size_t)M_ * DK_ * 4);
  float* obuf = (float*)alloc((size_t)M_ * E_ * 4);
  unsigned short* qt_g  = (unsigned short*)alloc((size_t)NBHC_ * 4096 * 2);
  unsigned short* s0t_g = (unsigned short*)alloc((size_t)NBHC_ * 8192 * 2);
  float* slocal = (float*)alloc((size_t)NBHC_ * 8192 * 4);
  float* expbc  = (float*)alloc((size_t)NBHC_ * 64 * 4);

  pack_wall_kernel<<<(NALL_ * E_) / 256, 256, 0, stream>>>(wq, wkg, wv, wg, wallt);
  pack_wout_kernel<<<(E_ * E_) / 256, 256, 0, stream>>>(wout, woutt);
  conv_silu_kernel<<<(M_ * E_) / 256, 256, 0, stream>>>(x, cw, xcb);
  gemm_bf16_wmma<<<dim3(NALL_ / 64, M_ / 128), 256, 0, stream>>>(
      xcb, wallt, qkvg, M_, NALL_, E_);
  gk_kernel<<<(M_ * DK_) / 256, 256, 0, stream>>>(qkvg, gkb, kbuf);
  gbias_kernel<<<(M_ * E_) / 256, 256, 0, stream>>>(qkvg, bg);
  gla_pass1_kernel<<<NBHC_, 256, 0, stream>>>(qkvg, gkb, kbuf, obuf,
                                              qt_g, slocal, expbc);
  gla_pass2_kernel<<<NBH_, 256, 0, stream>>>(slocal, expbc, s0t_g);
  gla_pass3_kernel<<<NBHC_, 256, 0, stream>>>(qt_g, s0t_g, obuf);
  post_kernel<<<M_, 256, 0, stream>>>(qkvg, kbuf, obuf, aug, preb);
  gemm_bf16_wmma<<<dim3(E_ / 64, M_ / 128), 256, 0, stream>>>(
      preb, woutt, out, M_, E_, E_);
}
